// Attention_67765993996440
// MI455X (gfx1250) — matrix operand order repaired and checked by reading, not compiled, not run on hardware
//
#include <hip/hip_runtime.h>

#define Tdim 2048
#define Ndim 2048
#define BHn  8
#define BM   128
#define BN   128
#define BK   32
#define LDA  40   // padded LDS row stride in halves (80B: 16B-aligned, conflict-spread)

typedef __attribute__((ext_vector_type(16))) _Float16 v16h;
typedef __attribute__((ext_vector_type(8)))  _Float16 v8h;
typedef __attribute__((ext_vector_type(4)))  _Float16 v4h;
typedef __attribute__((ext_vector_type(8)))  float    v8f;

struct H2 { _Float16 x, y; };

// ---- CDNA5 async global->LDS copy (ASYNCcnt-tracked, bypasses VGPRs) -------
// Generic pointers to __shared__ carry the LDS byte offset in their low 32 bits
// (LDS aperture: addr[63:32]=aperture, addr[31:0]=workgroup-relative offset).
__device__ __forceinline__ void async_b128(void* lds_dst, const void* gsrc) {
  unsigned ldsoff = (unsigned)(unsigned long long)lds_dst;
  asm volatile("global_load_async_to_lds_b128 %0, %1, off"
               :: "v"(ldsoff), "v"(gsrc) : "memory");
}
#define WAIT_ASYNC0() asm volatile("s_wait_asynccnt 0x0" ::: "memory")

__device__ __forceinline__ v16h frag_cat(v8h lo, v8h hi) {
  return __builtin_shufflevector(lo, hi, 0,1,2,3,4,5,6,7,8,9,10,11,12,13,14,15);
}

// A-matrix 16x32 f16 fragment from a row-major [16][LDA] LDS tile.
// ISA layout: lane<16 -> row=lane, K in {k0..k0+7, k0+16..k0+23}, k0 = (lane&16)?8:0
__device__ __forceinline__ v16h load_fragA(const _Float16* tile, int lane) {
  const int m  = lane & 15;
  const int k0 = (lane & 16) ? 8 : 0;
  const _Float16* p = tile + m * LDA + k0;
  v8h lo = *(const v8h*)(p);        // K = k0 .. k0+7
  v8h hi = *(const v8h*)(p + 16);   // K = k0+16 .. k0+23
  return frag_cat(lo, hi);
}

// B-matrix 32x16 f16 fragment from a row-major [16(n)][LDA(k)] LDS tile (B^T).
// ISA layout: lane<16 -> col n=lane, K in {k0..k0+7, k0+16..k0+23}, k0 = (lane&16)?8:0
__device__ __forceinline__ v16h load_fragB(const _Float16* tile, int lane) {
  const int n  = lane & 15;
  const int k0 = (lane & 16) ? 8 : 0;
  const _Float16* p = tile + n * LDA + k0;
  v8h lo = *(const v8h*)(p);
  v8h hi = *(const v8h*)(p + 16);
  return frag_cat(lo, hi);
}

__device__ __forceinline__ v8f wmma_f16(v16h a, v16h b, v8f c) {
  return __builtin_amdgcn_wmma_f32_16x16x32_f16(false, a, false, b, (short)0, c,
                                                false, false);
}

// ---------------- Pass 1a: RoPE(Q) -> f16 -----------------------------------
__global__ __launch_bounds__(256) void rope_kernel(const float* __restrict__ Q,
                                                   _Float16* __restrict__ qrh) {
  const size_t p = (size_t)blockIdx.x * blockDim.x + threadIdx.x;  // pair index
  const int halfN = Ndim / 2;
  const int n2 = (int)(p % halfN);
  const int t  = (int)((p / halfN) % Tdim);

  // theta = 2^16 -> freq = 2^(-n/128) * (1/2pi)
  float freq  = exp2f((float)(2 * n2) * (-1.0f / 128.0f)) * 0.15915494309189535f;
  float phase = (float)t * freq;
  phase -= floorf(phase);
  float sn, cs;
  __sincosf(phase * 6.283185307179586f, &sn, &cs);

  const float2 q = ((const float2*)Q)[p];
  H2 qo;
  qo.x = (_Float16)(q.x * cs - q.y * sn);
  qo.y = (_Float16)(q.y * cs + q.x * sn);
  ((H2*)qrh)[p] = qo;
}

// ---------------- Pass 1b: VT[n][t] = (f16) V[t][n] -------------------------
__global__ __launch_bounds__(256) void transpose_v(const float* __restrict__ V,
                                                   _Float16* __restrict__ vt) {
  const int n0 = blockIdx.x * 64;
  const int t0 = blockIdx.y * 64;
  const int bh = blockIdx.z;
  const float* Vb = V + (size_t)bh * Tdim * Ndim;
  _Float16* Tb = vt + (size_t)bh * (size_t)Ndim * Tdim;

  __shared__ __align__(16) _Float16 tile[64 * 72];   // 72-half rows (144B)
  const int tid = threadIdx.x;
  const int c4 = (tid & 15) * 4;
  const int r  = tid >> 4;                            // 0..15
#pragma unroll
  for (int i = 0; i < 4; ++i) {
    const int rr = r + i * 16;                        // t within tile
    float4 f = *(const float4*)(Vb + (size_t)(t0 + rr) * Ndim + n0 + c4);
    H2* d = (H2*)(tile + rr * 72 + c4);
    d[0] = { (_Float16)f.x, (_Float16)f.y };
    d[1] = { (_Float16)f.z, (_Float16)f.w };
  }
  __syncthreads();
#pragma unroll
  for (int i = 0; i < 4; ++i) {
    const int rr = r + i * 16;                        // n within tile
    v4h o;
#pragma unroll
    for (int j = 0; j < 4; ++j) o[j] = tile[(c4 + j) * 72 + rr];
    *(v4h*)(Tb + (size_t)(n0 + rr) * Tdim + t0 + c4) = o;
  }
}

// ---------------- Pass 2: scores = QR @ QR^T, strictly-causal masked --------
// Block tile 128x128, 8 waves (4 along M x 2 along N), wave tile 32x64,
// double-buffered LDS fed by async global->LDS copies, one barrier per K-step,
// no unroll so accumulators keep a single register home (no v_mov shuffling).
__global__ __launch_bounds__(256) void scores_kernel(const _Float16* __restrict__ qrh,
                                                     _Float16* __restrict__ sc) {
  const int sB = blockIdx.x * BN;
  const int tB = blockIdx.y * BM;
  if (sB >= tB + BM) return;                 // fully above diagonal: never read
  const int bh = blockIdx.z;
  const _Float16* Qh = qrh + (size_t)bh * Tdim * Ndim;
  _Float16* S = sc + (size_t)bh * Tdim * Tdim;

  __shared__ __align__(16) _Float16 As[2][BM * LDA];
  __shared__ __align__(16) _Float16 Bs[2][BN * LDA];

  const int tid  = threadIdx.x;
  const int lane = tid & 31;
  const int wave = tid >> 5;
  const int wm   = wave & 3;                 // 32-row slab
  const int wn   = wave >> 2;                // 64-col slab

  const int lr = tid >> 2;                   // 0..63
  const int lc = (tid & 3) * 8;              // 0,8,16,24
  const int so = lr * LDA + lc;

  const _Float16* gA0 = Qh + (size_t)(tB + lr) * Ndim + lc;
  const _Float16* gA1 = Qh + (size_t)(tB + lr + 64) * Ndim + lc;
  const _Float16* gB0 = Qh + (size_t)(sB + lr) * Ndim + lc;
  const _Float16* gB1 = Qh + (size_t)(sB + lr + 64) * Ndim + lc;

  v8f acc[2][4] = {};

  // preload tile 0 via async copies
  async_b128(As[0] + so, gA0);
  async_b128(As[0] + so + 64 * LDA, gA1);
  async_b128(Bs[0] + so, gB0);
  async_b128(Bs[0] + so + 64 * LDA, gB1);
  gA0 += BK; gA1 += BK; gB0 += BK; gB1 += BK;

  const int nsteps = Ndim / BK;
#pragma unroll 1
  for (int s = 0; s < nsteps; ++s) {
    const int cur = s & 1;
    // our async writes into buf[cur] must be complete before the barrier so
    // that every wave sees a fully-populated tile after it.
    WAIT_ASYNC0();
    __syncthreads();                         // also: prev readers of buf[cur^1] done
    if (s + 1 < nsteps) {                    // start next tile into the other buffer
      _Float16* dA = As[cur ^ 1];  _Float16* dB = Bs[cur ^ 1];
      async_b128(dA + so, gA0);
      async_b128(dA + so + 64 * LDA, gA1);
      async_b128(dB + so, gB0);
      async_b128(dB + so + 64 * LDA, gB1);
      gA0 += BK; gA1 += BK; gB0 += BK; gB1 += BK;
    }
    const _Float16* cA = As[cur];
    const _Float16* cB = Bs[cur];
    v16h a0 = load_fragA(cA + (wm * 32)      * LDA, lane);
    v16h a1 = load_fragA(cA + (wm * 32 + 16) * LDA, lane);
    v16h b0 = load_fragB(cB + (wn * 64)      * LDA, lane);
    v16h b1 = load_fragB(cB + (wn * 64 + 16) * LDA, lane);
    v16h b2 = load_fragB(cB + (wn * 64 + 32) * LDA, lane);
    v16h b3 = load_fragB(cB + (wn * 64 + 48) * LDA, lane);
    acc[0][0] = wmma_f16(a0, b0, acc[0][0]);
    acc[0][1] = wmma_f16(a0, b1, acc[0][1]);
    acc[0][2] = wmma_f16(a0, b2, acc[0][2]);
    acc[0][3] = wmma_f16(a0, b3, acc[0][3]);
    acc[1][0] = wmma_f16(a1, b0, acc[1][0]);
    acc[1][1] = wmma_f16(a1, b1, acc[1][1]);
    acc[1][2] = wmma_f16(a1, b2, acc[1][2]);
    acc[1][3] = wmma_f16(a1, b3, acc[1][3]);
  }

  const int rbase = (lane & 16) ? 8 : 0;
  const int col   = lane & 15;
#pragma unroll
  for (int i = 0; i < 2; ++i)
#pragma unroll
    for (int j = 0; j < 4; ++j) {
      const int scol = sB + wn * 64 + j * 16 + col;
#pragma unroll
      for (int e = 0; e < 8; ++e) {
        const int trow = tB + wm * 32 + i * 16 + rbase + e;
        const float v = (scol < trow) ? acc[i][j][e] : 0.0f;   // strict causal
        S[(size_t)trow * Tdim + scol] = (_Float16)v;
      }
    }
}

// ---------------- Pass 3: out = scores @ V (causal K-bound) -----------------
__global__ __launch_bounds__(256) void out_kernel(const _Float16* __restrict__ sc,
                                                  const _Float16* __restrict__ vt,
                                                  float* __restrict__ out) {
  const int nB = blockIdx.x * BN;
  const int tB = blockIdx.y * BM;
  const int bh = blockIdx.z;
  const _Float16* S  = sc + (size_t)bh * Tdim * Tdim;
  const _Float16* VT = vt + (size_t)bh * (size_t)Ndim * Tdim;   // [N][T]
  float* O = out + (size_t)bh * Tdim * Ndim;

  __shared__ __align__(16) _Float16 As[2][BM * LDA];
  __shared__ __align__(16) _Float16 Bs[2][BN * LDA];

  const int tid  = threadIdx.x;
  const int lane = tid & 31;
  const int wave = tid >> 5;
  const int wm   = wave & 3;
  const int wn   = wave >> 2;

  const int lr = tid >> 2;
  const int lc = (tid & 3) * 8;
  const int so = lr * LDA + lc;

  const _Float16* gA0 = S  + (size_t)(tB + lr) * Tdim + lc;
  const _Float16* gA1 = S  + (size_t)(tB + lr + 64) * Tdim + lc;
  const _Float16* gB0 = VT + (size_t)(nB + lr) * Tdim + lc;   // rows = n, cols = s
  const _Float16* gB1 = VT + (size_t)(nB + lr + 64) * Tdim + lc;

  v8f acc[2][4] = {};

  async_b128(As[0] + so, gA0);
  async_b128(As[0] + so + 64 * LDA, gA1);
  async_b128(Bs[0] + so, gB0);
  async_b128(Bs[0] + so + 64 * LDA, gB1);
  gA0 += BK; gA1 += BK; gB0 += BK; gB1 += BK;

  const int nsteps = (tB + BM) / BK;        // rows t need only s < tB+BM
#pragma unroll 1
  for (int s = 0; s < nsteps; ++s) {
    const int cur = s & 1;
    WAIT_ASYNC0();
    __syncthreads();
    if (s + 1 < nsteps) {
      _Float16* dA = As[cur ^ 1];  _Float16* dB = Bs[cur ^ 1];
      async_b128(dA + so, gA0);
      async_b128(dA + so + 64 * LDA, gA1);
      async_b128(dB + so, gB0);
      async_b128(dB + so + 64 * LDA, gB1);
      gA0 += BK; gA1 += BK; gB0 += BK; gB1 += BK;
    }
    const _Float16* cA = As[cur];
    const _Float16* cB = Bs[cur];
    v16h a0 = load_fragA(cA + (wm * 32)      * LDA, lane);
    v16h a1 = load_fragA(cA + (wm * 32 + 16) * LDA, lane);
    v16h b0 = load_fragB(cB + (wn * 64)      * LDA, lane);
    v16h b1 = load_fragB(cB + (wn * 64 + 16) * LDA, lane);
    v16h b2 = load_fragB(cB + (wn * 64 + 32) * LDA, lane);
    v16h b3 = load_fragB(cB + (wn * 64 + 48) * LDA, lane);
    acc[0][0] = wmma_f16(a0, b0, acc[0][0]);
    acc[0][1] = wmma_f16(a0, b1, acc[0][1]);
    acc[0][2] = wmma_f16(a0, b2, acc[0][2]);
    acc[0][3] = wmma_f16(a0, b3, acc[0][3]);
    acc[1][0] = wmma_f16(a1, b0, acc[1][0]);
    acc[1][1] = wmma_f16(a1, b1, acc[1][1]);
    acc[1][2] = wmma_f16(a1, b2, acc[1][2]);
    acc[1][3] = wmma_f16(a1, b3, acc[1][3]);
  }

  const int rbase = (lane & 16) ? 8 : 0;
  const int col   = lane & 15;
#pragma unroll
  for (int i = 0; i < 2; ++i)
#pragma unroll
    for (int j = 0; j < 4; ++j) {
#pragma unroll
      for (int e = 0; e < 8; ++e) {
        const int trow = tB + wm * 32 + i * 16 + rbase + e;
        const int ncol = nB + wn * 64 + j * 16 + col;
        O[(size_t)trow * Ndim + ncol] = acc[i][j][e];
      }
    }
}

extern "C" void kernel_launch(void* const* d_in, const int* in_sizes, int n_in,
                              void* d_out, int out_size, void* d_ws, size_t ws_size,
                              hipStream_t stream) {
  const float* Q = (const float*)d_in[0];
  const float* V = (const float*)d_in[1];
  float* out = (float*)d_out;

  const size_t perArr = (size_t)BHn * Tdim * Ndim;   // 33,554,432 halves = 64 MiB
  _Float16* qrh = (_Float16*)d_ws;
  _Float16* vt  = qrh + perArr;                      // [BH][N][T] f16
  _Float16* sc  = vt + perArr;                       // [BH][T][T] f16

  // Pass 1a: RoPE + downconvert Q (one thread per interleaved pair).
  const size_t pairs = perArr / 2;
  rope_kernel<<<dim3((unsigned)(pairs / 256)), dim3(256), 0, stream>>>(Q, qrh);

  // Pass 1b: transposed f16 copy of V.
  dim3 gt(Ndim / 64, Tdim / 64, BHn);
  transpose_v<<<gt, dim3(256), 0, stream>>>(V, vt);

  // Pass 2: causal scores GEMM.
  dim3 g2(Tdim / BN, Tdim / BM, BHn);
  scores_kernel<<<g2, dim3(256), 0, stream>>>(qrh, sc);

  // Pass 3: out = scores @ V with causal K-bound.
  dim3 g3(Ndim / BN, Tdim / BM, BHn);
  out_kernel<<<g3, dim3(256), 0, stream>>>(sc, vt, out);
}
